// FCITS_29326036697282
// MI455X (gfx1250) — compile-verified, matrix operand order checked by reading, not run on hardware
//
#include <hip/hip_runtime.h>
#include <hip/hip_bf16.h>
#include <math.h>

typedef __attribute__((ext_vector_type(16))) _Float16 v16h;
typedef __attribute__((ext_vector_type(8)))  _Float16 v8h;
typedef __attribute__((ext_vector_type(8)))  float    v8f;

#define EPS 1e-5f
#define S_PER_WG 16
#define NTHREADS 256
#define NWAVES   8

// padded activation row counts (one zero row each side)
#define H1_ROWS 34   // conv2 input positions -1..32
#define H2_ROWS 18   // conv3 input positions -1..16
#define H1_STRIDE (H1_ROWS * 64)   // 2176 halves per sample
#define H2_STRIDE (H2_ROWS * 32)   // 576 halves per sample

// ---- dynamic LDS layout ----------------------------------------------------
static constexpr int SMEM_FLOATS = S_PER_WG * 64 + 32 + 32 + 16 + 16;  // 1120
static constexpr int SMEM_HALVES = S_PER_WG * H1_STRIDE + S_PER_WG * H2_STRIDE +
                                   S_PER_WG * 128 + S_PER_WG * 112 +
                                   3 * 32 * 64 + 3 * 16 * 32 + 112 * 128;  // 69888
static constexpr int SMEM_BYTES  = SMEM_FLOATS * 4 + SMEM_HALVES * 2;      // 144256

__device__ __forceinline__ v8f wmma_f16(v16h a, v16h b, v8f c) {
    return __builtin_amdgcn_wmma_f32_16x16x32_f16(false, a, false, b, (short)0, c,
                                                  false, false);
}

// Pack two contiguous 8-half chunks (K = k0..k0+7 and k0+16..k0+23) into the
// ISA 16-bit A/B fragment register layout.
__device__ __forceinline__ v16h mk16(v8h lo, v8h hi) {
    v16h r;
#pragma unroll
    for (int i = 0; i < 8; ++i) { r[i] = lo[i]; r[i + 8] = hi[i]; }
    return r;
}

__device__ __forceinline__ v16h frag_ld(const _Float16* row, int k0) {
    return mk16(*(const v8h*)(row + k0), *(const v8h*)(row + k0 + 16));
}

__global__ __launch_bounds__(NTHREADS) void fcits_fused(
    const float* __restrict__ xg,
    const float* __restrict__ W1g, const float* __restrict__ b1g,
    const float* __restrict__ g1g, const float* __restrict__ be1g,
    const float* __restrict__ m1g, const float* __restrict__ v1g,
    const float* __restrict__ W2g, const float* __restrict__ b2g,
    const float* __restrict__ g2g, const float* __restrict__ be2g,
    const float* __restrict__ m2g, const float* __restrict__ v2g,
    const float* __restrict__ W3g, const float* __restrict__ b3g,
    const float* __restrict__ g3g, const float* __restrict__ be3g,
    const float* __restrict__ m3g, const float* __restrict__ v3g,
    const float* __restrict__ Wfcg, const float* __restrict__ Wclsg,
    const float* __restrict__ bclsg,
    float* __restrict__ logits, int N)
{
    extern __shared__ char smem[];
    float*    xs   = (float*)smem;            // [16][64]
    float*    sc2  = xs  + S_PER_WG * 64;
    float*    bi2  = sc2 + 32;
    float*    sc3  = bi2 + 32;
    float*    bi3  = sc3 + 16;
    _Float16* h1t  = (_Float16*)(bi3 + 16);   // [16][34 pos][64 cin], row p at (p+1)
    _Float16* h2t  = h1t + S_PER_WG * H1_STRIDE; // [16][18 pos][32 cin], row p at (p+1)
    _Float16* h3   = h2t + S_PER_WG * H2_STRIDE; // [16][128]  (ch*8+pos)
    _Float16* fcb  = h3  + S_PER_WG * 128;    // [16][112]
    _Float16* lw2d = fcb + S_PER_WG * 112;    // [3][32 co][64 cin]
    _Float16* lw3d = lw2d + 3 * 32 * 64;      // [3][16 co][32 cin]
    _Float16* lwfc = lw3d + 3 * 16 * 32;      // [112 n][128 k] (zero pad n>=100)

    const int tid   = threadIdx.x;
    const int wave  = tid >> 5;
    const int lane  = tid & 31;
    const int nloc  = lane & 15;
    const int mrow  = lane & 15;
    const int kbase = (lane < 16) ? 0 : 8;
    const int hi8   = (lane < 16) ? 0 : 8;
    const int wgBase = blockIdx.x * S_PER_WG;

    // ---- stage weights (N-major), BN folds, x, and zero pad rows ----------
    for (int i = tid; i < 3 * 32 * 64; i += NTHREADS) {
        int dk = i >> 11, rem = i & 2047, co = rem >> 6, cin = rem & 63;
        lw2d[i] = (_Float16)W2g[co * 192 + cin * 3 + dk];
    }
    for (int i = tid; i < 3 * 16 * 32; i += NTHREADS) {
        int dk = i >> 9, rem = i & 511, co = rem >> 5, cin = rem & 31;
        lw3d[i] = (_Float16)W3g[co * 96 + cin * 3 + dk];
    }
    for (int i = tid; i < 112 * 128; i += NTHREADS) {
        int n = i >> 7;
        lwfc[i] = (n < 100) ? (_Float16)Wfcg[i] : (_Float16)0.f;
    }
    // zero the padding rows of h1t (rows 0 and 33) and h2t (rows 0 and 17)
    for (int i = tid; i < S_PER_WG * 2 * 64; i += NTHREADS) {
        int s = i >> 7, r = (i >> 6) & 1, c = i & 63;
        h1t[s * H1_STRIDE + (r ? (H1_ROWS - 1) : 0) * 64 + c] = (_Float16)0.f;
    }
    for (int i = tid; i < S_PER_WG * 2 * 32; i += NTHREADS) {
        int s = i >> 6, r = (i >> 5) & 1, c = i & 31;
        h2t[s * H2_STRIDE + (r ? (H2_ROWS - 1) : 0) * 32 + c] = (_Float16)0.f;
    }
    if (tid < 32) {
        int c = tid;
        float sc = g2g[c] * rsqrtf(v2g[c] + EPS);
        sc2[c] = sc;
        bi2[c] = (b2g[c] - m2g[c]) * sc + be2g[c];
    } else if (tid < 48) {
        int c = tid - 32;
        float sc = g3g[c] * rsqrtf(v3g[c] + EPS);
        sc3[c] = sc;
        bi3[c] = (b3g[c] - m3g[c]) * sc + be3g[c];
    }
    for (int i = tid; i < S_PER_WG * 64; i += NTHREADS) {
        long g = (long)wgBase * 64 + i;
        xs[i] = (g < (long)N * 64) ? xg[g] : 0.f;
    }
    __syncthreads();

    // ---- conv1 (Cin=1,K=3) + BN + ReLU + pool2 : VALU, write [pos][cin] ---
    for (int idx = tid; idx < S_PER_WG * 64; idx += NTHREADS) {
        int s = idx >> 6, c = idx & 63;
        float w0 = W1g[c * 3 + 0], w1 = W1g[c * 3 + 1], w2 = W1g[c * 3 + 2];
        float sc = g1g[c] * rsqrtf(v1g[c] + EPS);
        float bi = (b1g[c] - m1g[c]) * sc + be1g[c];
        const float* xp = xs + s * 64;
        _Float16* hp = h1t + s * H1_STRIDE + 64 + c;   // row p=0 is at offset 64
        for (int q = 0; q < 32; ++q) {
            int p0 = 2 * q, p1 = 2 * q + 1;
            float a0 = (p0 > 0 ? xp[p0 - 1] : 0.f) * w0 + xp[p0] * w1 + xp[p0 + 1] * w2;
            float a1 = xp[p1 - 1] * w0 + xp[p1] * w1 + (p1 < 63 ? xp[p1 + 1] : 0.f) * w2;
            a0 = fmaxf(sc * a0 + bi, 0.f);
            a1 = fmaxf(sc * a1 + bi, 0.f);
            hp[q * 64] = (_Float16)fmaxf(a0, a1);
        }
    }
    __syncthreads();

    // ---- conv2: 3 shifted GEMMs, K=64 each : WMMA (unconditional loads) ---
    {
        v16h b2f[3][2][2];   // [dk][kt][nt], sample-invariant
#pragma unroll
        for (int dk = 0; dk < 3; ++dk)
#pragma unroll
            for (int kt = 0; kt < 2; ++kt) {
                int k0 = kt * 32 + kbase;
                b2f[dk][kt][0] = frag_ld(lw2d + dk * 2048 + nloc * 64, k0);
                b2f[dk][kt][1] = frag_ld(lw2d + dk * 2048 + (16 + nloc) * 64, k0);
            }

        for (int s = wave; s < S_PER_WG; s += NWAVES) {
            const _Float16* h1s = h1t + s * H1_STRIDE;
            for (int mt = 0; mt < 2; ++mt) {
                int pos = mt * 16 + mrow;
                v8f acc0 = {}, acc1 = {};
#pragma unroll
                for (int dk = 0; dk < 3; ++dk) {
                    // conv position p = pos-1+dk lives at padded row p+1 = pos+dk
                    const _Float16* rowp = h1s + (pos + dk) * 64;
#pragma unroll
                    for (int kt = 0; kt < 2; ++kt) {
                        int k0 = kt * 32 + kbase;
                        v16h a = frag_ld(rowp, k0);
                        acc0 = wmma_f16(a, b2f[dk][kt][0], acc0);
                        acc1 = wmma_f16(a, b2f[dk][kt][1], acc1);
                    }
                }
                int off = mt * 16 + hi8;
#pragma unroll
                for (int nt = 0; nt < 2; ++nt) {
                    int ch = nt * 16 + nloc;
                    float sc = sc2[ch], bi = bi2[ch];
                    const v8f& acc = nt ? acc1 : acc0;
#pragma unroll
                    for (int r = 0; r < 8; r += 2) {
                        float y0 = fmaxf(sc * acc[r]     + bi, 0.f);
                        float y1 = fmaxf(sc * acc[r + 1] + bi, 0.f);
                        // pooled pos pp -> padded row pp+1
                        h2t[s * H2_STRIDE + (((off + r) >> 1) + 1) * 32 + ch] =
                            (_Float16)fmaxf(y0, y1);
                    }
                }
            }
        }
    }
    __syncthreads();

    // ---- conv3: 3 shifted GEMMs, K=32 each : WMMA -------------------------
    {
        v16h b3f[3];
#pragma unroll
        for (int dk = 0; dk < 3; ++dk)
            b3f[dk] = frag_ld(lw3d + dk * 512 + nloc * 32, kbase);

        for (int s = wave; s < S_PER_WG; s += NWAVES) {
            const _Float16* h2s = h2t + s * H2_STRIDE;
            v8f acc = {};
#pragma unroll
            for (int dk = 0; dk < 3; ++dk) {
                const _Float16* rowp = h2s + (mrow + dk) * 32;  // p+1 = mrow+dk
                acc = wmma_f16(frag_ld(rowp, kbase), b3f[dk], acc);
            }
            int ch = nloc;
            float sc = sc3[ch], bi = bi3[ch];
#pragma unroll
            for (int r = 0; r < 8; r += 2) {
                float y0 = fmaxf(sc * acc[r]     + bi, 0.f);
                float y1 = fmaxf(sc * acc[r + 1] + bi, 0.f);
                h3[s * 128 + ch * 8 + ((hi8 + r) >> 1)] = (_Float16)fmaxf(y0, y1);
            }
        }
    }
    __syncthreads();

    // ---- FC: [16,128] x [128,112(pad)] : WMMA, ReLU -----------------------
    if (wave < 7) {
        const _Float16* arow = h3 + mrow * 128;
        const _Float16* brow = lwfc + (wave * 16 + nloc) * 128;
        v8f acc = {};
#pragma unroll
        for (int kt = 0; kt < 4; ++kt) {
            int k0 = kt * 32 + kbase;
            acc = wmma_f16(frag_ld(arow, k0), frag_ld(brow, k0), acc);
        }
        int f = wave * 16 + nloc;
#pragma unroll
        for (int r = 0; r < 8; ++r)
            fcb[(hi8 + r) * 112 + f] = (_Float16)fmaxf(acc[r], 0.f);
    }
    __syncthreads();

    // ---- classifier [16,100]x[100,6]+b : VALU -----------------------------
    if (tid < S_PER_WG * 6) {
        int s = tid / 6, j = tid - 6 * s;
        float sum = bclsg[j];
        const _Float16* hv = fcb + s * 112;
        const float* wr = Wclsg + j * 100;
        for (int k = 0; k < 100; ++k) sum += (float)hv[k] * wr[k];
        int g = wgBase + s;
        if (g < N) logits[g * 6 + j] = sum;
    }
}

// ---- LICell scan -----------------------------------------------------------
__global__ __launch_bounds__(256) void li_scan(const float* __restrict__ logits,
                                               const float* __restrict__ tau,
                                               float* __restrict__ out,
                                               int B, int T)
{
    int t = blockIdx.x * blockDim.x + threadIdx.x;
    if (t >= B * 6) return;
    int b = t / 6, j = t - 6 * b;
    float alpha = expf(-1.f / fabsf(tau[j]));
    float omc = 1.f - alpha;
    float u = 0.f;
    for (int k = 0; k < T; ++k) {
        int idx = (k * B + b) * 6 + j;
        u = u * alpha + logits[idx] * omc;
        out[idx] = u;
    }
}

extern "C" void kernel_launch(void* const* d_in, const int* in_sizes, int n_in,
                              void* d_out, int out_size, void* d_ws, size_t ws_size,
                              hipStream_t stream) {
    const float* x    = (const float*)d_in[0];
    const float* W1   = (const float*)d_in[1];
    const float* b1   = (const float*)d_in[2];
    const float* g1   = (const float*)d_in[3];
    const float* be1  = (const float*)d_in[4];
    const float* m1   = (const float*)d_in[5];
    const float* v1   = (const float*)d_in[6];
    const float* W2   = (const float*)d_in[7];
    const float* b2   = (const float*)d_in[8];
    const float* g2   = (const float*)d_in[9];
    const float* be2  = (const float*)d_in[10];
    const float* m2   = (const float*)d_in[11];
    const float* v2   = (const float*)d_in[12];
    const float* W3   = (const float*)d_in[13];
    const float* b3   = (const float*)d_in[14];
    const float* g3   = (const float*)d_in[15];
    const float* be3  = (const float*)d_in[16];
    const float* m3   = (const float*)d_in[17];
    const float* v3   = (const float*)d_in[18];
    const float* Wfc  = (const float*)d_in[19];
    const float* Wcls = (const float*)d_in[20];
    const float* bcls = (const float*)d_in[21];
    const float* tau  = (const float*)d_in[22];

    const int N = in_sizes[0] / 64;   // 28160
    const int T = 110;
    const int B = N / T;              // 256

    float* logits = (float*)d_ws;

    int nwg = (N + S_PER_WG - 1) / S_PER_WG;
    fcits_fused<<<nwg, NTHREADS, SMEM_BYTES, stream>>>(
        x, W1, b1, g1, be1, m1, v1,
        W2, b2, g2, be2, m2, v2,
        W3, b3, g3, be3, m3, v3,
        Wfc, Wcls, bcls, logits, N);

    li_scan<<<(B * 6 + 255) / 256, 256, 0, stream>>>(logits, tau, (float*)d_out, B, T);
}
